// AttentionHead_44504451121596
// MI455X (gfx1250) — matrix operand order repaired and checked by reading, not compiled, not run on hardware
//
#include <hip/hip_runtime.h>

typedef __bf16 bf16_t;
typedef __attribute__((ext_vector_type(16))) __bf16 v16bf;
typedef __attribute__((ext_vector_type(8)))  float  v8f;

constexpr int B_    = 8;
constexpr int S_    = 2048;
constexpr int E_    = 768;
constexpr int I_    = 64;
constexpr int ROWS_ = B_ * S_;          // 16384 flattened (b,s) rows
constexpr int LSTR1 = 66;               // kernel-1 LDS stride (33 dwords)
constexpr int LSTR  = 72;               // kernel-2 LDS stride (144B rows, 16B aligned)

// ---------------------------------------------------------------------------
// WMMA fragment loaders (CDNA5 wave32 layouts, CDNA5 ISA §7.12.2)
// ---------------------------------------------------------------------------

// A-matrix 16x32 bf16: lanes 0-15 -> M=lane, K {0..7,16..23}; lanes 16-31 -> K+8.
__device__ inline v16bf load_frag_a(const bf16_t* base, int row_base, int stride, int k_base) {
  const int lane = threadIdx.x & 31;
  const bf16_t* p = base + (row_base + (lane & 15)) * stride + k_base + ((lane >> 4) ? 8 : 0);
  v16bf a;
#pragma unroll
  for (int v = 0; v < 8; ++v) {
    const int k = (v < 4) ? (2 * v) : (16 + 2 * (v - 4));
    a[2 * v]     = p[k];
    a[2 * v + 1] = p[k + 1];
  }
  return a;
}

// B-matrix 32x16 bf16: column N = lane&15 (stride sn), K = (lane>=16 ? 8:0)+{0..7,16..23} (stride sk).
__device__ inline v16bf load_frag_b(const bf16_t* base, int n_base, int sn, int k_base, int sk) {
  const int lane = threadIdx.x & 31;
  const bf16_t* p = base + (n_base + (lane & 15)) * sn + (k_base + ((lane >> 4) << 3)) * sk;
  v16bf b;
#pragma unroll
  for (int c = 0; c < 16; ++c) b[c] = p[(c < 8 ? c : c + 8) * sk];
  return b;
}

__device__ inline v8f wmma_bf16(v16bf a, v16bf b, v8f c) {
  return __builtin_amdgcn_wmma_f32_16x16x32_bf16(false, a, false, b, (short)0, c, false, false);
}

// low 32 bits of a flat LDS address = LDS byte offset (ISA §10.2 aperture calc)
__device__ inline uint32_t lds_off(const void* p) { return (uint32_t)(uintptr_t)p; }

#define WAIT_ASYNC_0() asm volatile("s_wait_asynccnt 0x0" ::: "memory")
#define WAIT_ASYNC_4() asm volatile("s_wait_asynccnt 0x4" ::: "memory")

// One 16B chunk: ASYNCcnt-tracked DMA into LDS, no VGPR staging.
__device__ inline void async_b128(uint32_t lds_byte, uint32_t voff_byte, uint64_t sbase) {
  asm volatile("global_load_async_to_lds_b128 %0, %1, %2"
               :: "v"(lds_byte), "v"(voff_byte), "s"(sbase) : "memory");
}

// Copy one 64-row x 128B K tile and V tile (2 chunks each per thread => 4 async ops)
__device__ inline void async_tile_kv(uint64_t kbase, uint64_t vbase, int row0,
                                     uint32_t ldsK_b, uint32_t ldsV_b) {
  const int tid = threadIdx.x;
#pragma unroll
  for (int j = 0; j < 2; ++j) {
    const int chunk = tid + j * 256;          // 512 chunks per matrix
    const int r = chunk >> 3, c = chunk & 7;  // 8 x 16B per 128B row
    const uint32_t voff = (uint32_t)(row0 + r) * 128u + (uint32_t)c * 16u;
    const uint32_t lo   = (uint32_t)r * (LSTR * 2) + (uint32_t)c * 16u;
    async_b128(ldsK_b + lo, voff, kbase);
    async_b128(ldsV_b + lo, voff, vbase);
  }
}

// ---------------------------------------------------------------------------
// Kernel 1: fused Q/K/V projection (emb @ W^T + bias), bf16 out to workspace
// grid = 256 (64 rows each), block = 256 (8 waves)
// ---------------------------------------------------------------------------
__global__ __launch_bounds__(256) void qkv_proj(
    const float* __restrict__ emb,
    const float* __restrict__ Wq, const float* __restrict__ bq,
    const float* __restrict__ Wk, const float* __restrict__ bk,
    const float* __restrict__ Wv, const float* __restrict__ bv,
    bf16_t* __restrict__ qkv_ws)            // [3][ROWS_][I_] bf16
{
  __shared__ bf16_t ldsA[64 * LSTR1];       // emb chunk  (64 rows x 64 k)
  __shared__ bf16_t ldsB[192 * LSTR1];      // weight chunk (Wq|Wk|Wv x 64 k)

  const int tid  = threadIdx.x;
  const int wave = tid >> 5;
  const int lane = tid & 31;
  const int row_base = blockIdx.x * 64;

  const int m_tile = wave & 3;              // 4 M-tiles of 16 rows
  const int n_half = wave >> 2;             // each half of waves covers 6 N-tiles

  v8f acc[6];
#pragma unroll
  for (int t = 0; t < 6; ++t) acc[t] = (v8f){};

  for (int kc = 0; kc < E_; kc += 64) {
    for (int idx = tid; idx < 64 * 64; idx += 256) {
      const int r = idx >> 6, c = idx & 63;
      ldsA[r * LSTR1 + c] = (bf16_t)emb[(size_t)(row_base + r) * E_ + kc + c];
    }
    for (int idx = tid; idx < 192 * 64; idx += 256) {
      const int r = idx >> 6, c = idx & 63;
      const float* W = (r < 64) ? Wq : (r < 128) ? Wk : Wv;
      ldsB[r * LSTR1 + c] = (bf16_t)W[(size_t)(r & 63) * E_ + kc + c];
    }
    __syncthreads();

    v16bf a0 = load_frag_a(ldsA, m_tile * 16, LSTR1, 0);
    v16bf a1 = load_frag_a(ldsA, m_tile * 16, LSTR1, 32);
#pragma unroll
    for (int t = 0; t < 6; ++t) {
      const int n_tile = n_half * 6 + t;
      v16bf b0 = load_frag_b(ldsB, n_tile * 16, LSTR1, 0, 1);
      v16bf b1 = load_frag_b(ldsB, n_tile * 16, LSTR1, 32, 1);
      acc[t] = wmma_bf16(a0, b0, acc[t]);
      acc[t] = wmma_bf16(a1, b1, acc[t]);
    }
    __syncthreads();
  }

  // bias + store bf16 (C/D layout: lane half h -> M += 8h, vgpr r -> row, lane&15 -> col)
  const int h  = lane >> 4;
  const int cl = lane & 15;
#pragma unroll
  for (int t = 0; t < 6; ++t) {
    const int n_tile = n_half * 6 + t;
    const int mat    = n_tile >> 2;                 // 0=Q 1=K 2=V
    const int ncol   = (n_tile & 3) * 16 + cl;
    const float* bias = (mat == 0) ? bq : (mat == 1) ? bk : bv;
    const float bb = bias[ncol];
    bf16_t* dst = qkv_ws + (size_t)mat * ROWS_ * I_;
#pragma unroll
    for (int r = 0; r < 8; ++r) {
      const int m = m_tile * 16 + r + 8 * h;
      dst[(size_t)(row_base + m) * I_ + ncol] = (bf16_t)(acc[t][r] + bb);
    }
  }
}

// ---------------------------------------------------------------------------
// Kernel 2: flash attention with ASYNC double-buffered K/V tiles.
// grid = 8 batches x 16 query-blocks(128 rows), block = 256 (8 waves)
// ---------------------------------------------------------------------------
__global__ __launch_bounds__(256) void flash_attn(
    const bf16_t* __restrict__ qkv, float* __restrict__ out)
{
  // Q block shares storage with per-wave P staging (each wave only touches
  // its own 16-row slice after fragment extraction).
  __shared__ __align__(16) bf16_t ldsQP[128 * LSTR];
  __shared__ __align__(16) bf16_t ldsK[2][64 * LSTR];
  __shared__ __align__(16) bf16_t ldsV[2][64 * LSTR];

  const int tid  = threadIdx.x;
  const int wave = tid >> 5;
  const int lane = tid & 31;
  const int b    = blockIdx.x >> 4;
  const int qb   = blockIdx.x & 15;
  const size_t qrow0 = (size_t)b * S_ + (size_t)qb * 128;

  const bf16_t* Qw = qkv;
  const bf16_t* Kw = qkv + (size_t)1 * ROWS_ * I_;
  const bf16_t* Vw = qkv + (size_t)2 * ROWS_ * I_;

  const uint64_t qbase = (uint64_t)(uintptr_t)(Qw + qrow0 * I_);
  const uint64_t kbase = (uint64_t)(uintptr_t)(Kw + (size_t)b * S_ * I_);
  const uint64_t vbase = (uint64_t)(uintptr_t)(Vw + (size_t)b * S_ * I_);
  const uint32_t ldsQP_b = lds_off(ldsQP);
  const uint32_t ldsK_b[2] = { lds_off(&ldsK[0][0]), lds_off(&ldsK[1][0]) };
  const uint32_t ldsV_b[2] = { lds_off(&ldsV[0][0]), lds_off(&ldsV[1][0]) };

  // async copy Q block (128 rows x 128B = 1024 chunks, 4 per thread)
#pragma unroll
  for (int j = 0; j < 4; ++j) {
    const int chunk = tid + j * 256;
    const int r = chunk >> 3, c = chunk & 7;
    async_b128(ldsQP_b + (uint32_t)r * (LSTR * 2) + (uint32_t)c * 16u,
               (uint32_t)r * 128u + (uint32_t)c * 16u, qbase);
  }
  // prefetch K/V tile 0 behind the Q copy
  async_tile_kv(kbase, vbase, 0, ldsK_b[0], ldsV_b[0]);
  WAIT_ASYNC_4();                // Q done (tile0 may still be in flight)
  __syncthreads();

  v16bf qf0 = load_frag_a(ldsQP, wave * 16, LSTR, 0);
  v16bf qf1 = load_frag_a(ldsQP, wave * 16, LSTR, 32);
  bf16_t* ldsP_w = &ldsQP[wave * 16 * LSTR];     // reuse Q slice for P staging

  v8f oacc[4];
#pragma unroll
  for (int f = 0; f < 4; ++f) oacc[f] = (v8f){};
  float mrow[8], lrow[8];
#pragma unroll
  for (int r = 0; r < 8; ++r) { mrow[r] = -__builtin_inff(); lrow[r] = 0.f; }

  const float scale = 0.036084391824351615f;      // 1/sqrt(768)

  for (int kb = 0; kb < S_; kb += 64) {
    const int buf = (kb >> 6) & 1;
    if (kb + 64 < S_) {
      // prefetch next tile into the buffer everyone finished reading last iter
      async_tile_kv(kbase, vbase, kb + 64, ldsK_b[buf ^ 1], ldsV_b[buf ^ 1]);
      WAIT_ASYNC_4();            // retire current tile, leave prefetch in flight
    } else {
      WAIT_ASYNC_0();
    }
    __syncthreads();             // publish current tile across waves

    const bf16_t* tK = &ldsK[buf][0];
    const bf16_t* tV = &ldsV[buf][0];

    // S = Q @ K^T for 4 key sub-tiles
    v8f s[4];
#pragma unroll
    for (int kt = 0; kt < 4; ++kt) {
      v8f a = (v8f){};
      v16bf bk0 = load_frag_b(tK, kt * 16, LSTR, 0, 1);
      v16bf bk1 = load_frag_b(tK, kt * 16, LSTR, 32, 1);
      a = wmma_bf16(qf0, bk0, a);
      a = wmma_bf16(qf1, bk1, a);
      s[kt] = a;
    }

    // online softmax: new row max over these 64 keys
    float mnew[8];
#pragma unroll
    for (int r = 0; r < 8; ++r) {
      float mx = mrow[r];
#pragma unroll
      for (int kt = 0; kt < 4; ++kt) mx = fmaxf(mx, s[kt][r] * scale);
      mnew[r] = mx;
    }
#pragma unroll
    for (int off = 1; off < 16; off <<= 1)
#pragma unroll
      for (int r = 0; r < 8; ++r)
        mnew[r] = fmaxf(mnew[r], __shfl_xor(mnew[r], off, 32));

    float corr[8];
#pragma unroll
    for (int r = 0; r < 8; ++r) {
      corr[r] = __expf(mrow[r] - mnew[r]);
      mrow[r] = mnew[r];
      lrow[r] *= corr[r];
    }
#pragma unroll
    for (int f = 0; f < 4; ++f)
#pragma unroll
      for (int r = 0; r < 8; ++r) oacc[f][r] *= corr[r];

    // P = exp(S*scale - m), stage to own LDS slice (C-layout -> A-frag reshape)
    const int h  = lane >> 4;
    const int cl = lane & 15;
#pragma unroll
    for (int kt = 0; kt < 4; ++kt)
#pragma unroll
      for (int r = 0; r < 8; ++r) {
        const float p = __expf(s[kt][r] * scale - mnew[r]);
        lrow[r] += p;                               // per-lane partial row sum
        ldsP_w[(r + 8 * h) * LSTR + kt * 16 + cl] = (bf16_t)p;
      }

    // O += P @ V  (two K=32 steps over the 64 keys)
#pragma unroll
    for (int ktp = 0; ktp < 2; ++ktp) {
      v16bf pf = load_frag_a(ldsP_w, 0, LSTR, ktp * 32);
#pragma unroll
      for (int f = 0; f < 4; ++f) {
        v16bf vf = load_frag_b(tV, f * 16, 1, ktp * 32, LSTR); // n=feature, K=key
        oacc[f] = wmma_bf16(pf, vf, oacc[f]);
      }
    }
    __syncthreads();             // all waves done with this tile before overwrite
  }

  // reduce row sums across the 16-lane half, normalize, store fp32
#pragma unroll
  for (int off = 1; off < 16; off <<= 1)
#pragma unroll
    for (int r = 0; r < 8; ++r) lrow[r] += __shfl_xor(lrow[r], off, 32);

  const int h  = lane >> 4;
  const int cl = lane & 15;
#pragma unroll
  for (int f = 0; f < 4; ++f)
#pragma unroll
    for (int r = 0; r < 8; ++r) {
      const int m = wave * 16 + r + 8 * h;
      out[(qrow0 + m) * I_ + f * 16 + cl] = oacc[f][r] / lrow[r];
    }
}

// ---------------------------------------------------------------------------
extern "C" void kernel_launch(void* const* d_in, const int* in_sizes, int n_in,
                              void* d_out, int out_size, void* d_ws, size_t ws_size,
                              hipStream_t stream) {
  (void)in_sizes; (void)n_in; (void)out_size; (void)ws_size;
  const float* emb = (const float*)d_in[0];
  const float* Wq  = (const float*)d_in[1];
  const float* bq  = (const float*)d_in[2];
  const float* Wk  = (const float*)d_in[3];
  const float* bk  = (const float*)d_in[4];
  const float* Wv  = (const float*)d_in[5];
  const float* bv  = (const float*)d_in[6];
  bf16_t* qkv_ws = (bf16_t*)d_ws;                     // 3 * 16384 * 64 * 2B = 6 MB
  float*  out    = (float*)d_out;

  qkv_proj<<<ROWS_ / 64, 256, 0, stream>>>(emb, Wq, bq, Wk, bk, Wv, bv, qkv_ws);
  flash_attn<<<(B_ * S_) / 128, 256, 0, stream>>>(qkv_ws, out);
}
